// Attention_78812649881930
// MI455X (gfx1250) — 8-bit matrix operand order repaired and checked by reading, compiled, not run on hardware
//
#include <hip/hip_runtime.h>
#include <cstdint>


typedef int v8i __attribute__((ext_vector_type(8)));

#define BSZ   8
#define NSEQ  784
#define CDIM  768
#define HN    12
#define DH    64
#define MROWS (BSZ*NSEQ)
#define NQKV  (3*CDIM)
#define NPAD  832
#define MB_X  (MROWS/64)
#define NB_Q  (NQKV/128)
#define NB_P  (CDIM/128)
#define NT_S  (NSEQ/16)

__device__ __forceinline__ v8i load_a16x64(const int8_t* A, int lda, int k0, int lane) {
  const int half = (lane >> 4) & 1, m = lane & 15;
  const int8_t* p = A + (size_t)m * lda + k0 + half * 16;
  v8i a; int4 t;
  t = *(const int4*)(p     ); a[0] = t.x; a[1] = t.y; a[2] = t.z; a[3] = t.w;
  t = *(const int4*)(p + 32); a[4] = t.x; a[5] = t.y; a[6] = t.z; a[7] = t.w;
  return a;
}
__device__ __forceinline__ v8i load_b64x16(const int8_t* Brows, int ldb, int k0, int lane) {
  const int half = (lane >> 4) & 1, n = lane & 15;
  const int8_t* p = Brows + (size_t)n * ldb + k0 + half * 16;
  v8i b; int4 t;
  t = *(const int4*)(p     ); b[0] = t.x; b[1] = t.y; b[2] = t.z; b[3] = t.w;
  t = *(const int4*)(p + 32); b[4] = t.x; b[5] = t.y; b[6] = t.z; b[7] = t.w;
  return b;
}

__device__ __forceinline__ int clip_round(float v, float lo, float hi) {
  float r = rintf(v);
  r = fmaxf(lo, fminf(hi, r));
  return (int)r;
}

__device__ __forceinline__ void async_g2l_b128(uint32_t lds_off, const int8_t* gp) {
  asm volatile("global_load_async_to_lds_b128 %0, %1, off"
               :: "v"(lds_off), "v"(gp) : "memory");
}
__device__ __forceinline__ void wait_async0() {
  asm volatile("s_wait_asynccnt 0x0" ::: "memory");
}


__global__ void k_init(float* g) {
  if (threadIdx.x < 16) ((unsigned*)g)[threadIdx.x] = 0u;
}

__global__ void k_quant_x(const float* __restrict__ x, const float* __restrict__ sx,
                          int8_t* __restrict__ x8) {
  size_t i = (size_t)blockIdx.x * 256 + threadIdx.x;
  float inv = 1.0f / sx[0];
  x8[i] = (int8_t)clip_round(x[i] * inv, -128.f, 127.f);
}

__global__ void k_quant_w(const float* __restrict__ Wq, const float* __restrict__ Wp,
                          int8_t* __restrict__ wq8, int8_t* __restrict__ wp8,
                          float* __restrict__ wqs, float* __restrict__ wps) {
  __shared__ float red[256];
  int r = blockIdx.x, t = threadIdx.x;
  const float* W; int8_t* o8; float* os;
  if (r < NQKV) { W = Wq + (size_t)r * CDIM; o8 = wq8 + (size_t)r * CDIM; os = &wqs[r]; }
  else { int rr = r - NQKV; W = Wp + (size_t)rr * CDIM; o8 = wp8 + (size_t)rr * CDIM; os = &wps[rr]; }
  float m = 0.f;
  for (int c = t; c < CDIM; c += 256) m = fmaxf(m, fabsf(W[c]));
  red[t] = m; __syncthreads();
  for (int s = 128; s > 0; s >>= 1) { if (t < s) red[t] = fmaxf(red[t], red[t + s]); __syncthreads(); }
  float ws = fmaxf(red[0] / 127.0f, 1e-30f);
  if (t == 0) *os = ws;
  float inv = 1.0f / ws;
  for (int c = t; c < CDIM; c += 256) o8[c] = (int8_t)clip_round(W[c] * inv, -128.f, 127.f);
}

template <int MODE>
__global__ void __launch_bounds__(128)
k_gemm_i8(const int8_t* __restrict__ A, const int8_t* __restrict__ W,
          int NB, int Ncols, int K,
          const float* __restrict__ wscale, const float* __restrict__ sx,
          float* __restrict__ g, int8_t* __restrict__ out8,
          float* __restrict__ outf, const float* __restrict__ bias) {
  __shared__ __align__(16) int8_t bufA[2][64 * 64];
  int t = threadIdx.x, lane = t & 31, wv = t >> 5;
  int mb = blockIdx.x / NB, nb = blockIdx.x % NB;
  const int8_t* Abase = A + (size_t)mb * 64 * K;
  const int8_t* W0 = W + (size_t)(nb * 128 + wv * 32) * K;

  #pragma unroll
  for (int s = 0; s < 2; s++) {
    int c = t + 128 * s;
    async_g2l_b128((uint32_t)(uintptr_t)&bufA[0][c * 16],
                   Abase + (size_t)(c >> 2) * K + (c & 3) * 16);
  }
  wait_async0();
  __syncthreads();

  v8i zero = {};
  v8i acc[8];
  #pragma unroll
  for (int i = 0; i < 8; i++) acc[i] = zero;

  const int nsteps = K / 64;
  for (int ks = 0; ks < nsteps; ks++) {
    int cur = ks & 1;
    if (ks + 1 < nsteps) {
      int k0n = (ks + 1) * 64;
      #pragma unroll
      for (int s = 0; s < 2; s++) {
        int c = t + 128 * s;
        async_g2l_b128((uint32_t)(uintptr_t)&bufA[cur ^ 1][c * 16],
                       Abase + (size_t)(c >> 2) * K + k0n + (c & 3) * 16);
      }
    }
    int k0 = ks * 64;
    v8i b0 = load_b64x16(W0, K, k0, lane);
    v8i b1 = load_b64x16(W0 + (size_t)16 * K, K, k0, lane);
    #pragma unroll
    for (int i = 0; i < 4; i++) {
      v8i a = load_a16x64((const int8_t*)&bufA[cur][i * 16 * 64], 64, 0, lane);
      acc[i * 2 + 0] = __builtin_amdgcn_wmma_i32_16x16x64_iu8(true, a, true, b0, acc[i * 2 + 0], false, false);
      acc[i * 2 + 1] = __builtin_amdgcn_wmma_i32_16x16x64_iu8(true, a, true, b1, acc[i * 2 + 1], false, false);
    }
    wait_async0();
    __syncthreads();
  }

  int half = (lane >> 4) & 1, cl = lane & 15;
  if (MODE == 0) {
    float sxv = sx[0];
    float m = 0.f;
    #pragma unroll
    for (int j = 0; j < 2; j++) {
      int col = nb * 128 + wv * 32 + j * 16 + cl;
      float wsx = wscale[col] * sxv;
      for (int i = 0; i < 4; i++)
        for (int v = 0; v < 8; v++)
          m = fmaxf(m, fabsf((float)acc[i * 2 + j][v] * wsx));
    }
    atomicMax((unsigned*)&g[0], __float_as_uint(m));
  } else if (MODE == 1) {
    float sxv = sx[0];
    float inv_s1 = 1.0f / g[1];
    #pragma unroll
    for (int j = 0; j < 2; j++) {
      int col = nb * 128 + wv * 32 + j * 16 + cl;
      float f = wscale[col] * sxv * inv_s1;
      for (int i = 0; i < 4; i++)
        for (int v = 0; v < 8; v++) {
          int row = mb * 64 + i * 16 + v + 8 * half;
          out8[(size_t)row * Ncols + col] =
              (int8_t)clip_round((float)acc[i * 2 + j][v] * f, -128.f, 127.f);
        }
    }
  } else {
    float m = 0.f;
    #pragma unroll
    for (int j = 0; j < 2; j++) {
      int col = nb * 128 + wv * 32 + j * 16 + cl;
      float os = wscale[col] * g[8];
      float bi = rintf(bias[col] / os);
      for (int i = 0; i < 4; i++)
        for (int v = 0; v < 8; v++) {
          int row = mb * 64 + i * 16 + v + 8 * half;
          float f = ((float)acc[i * 2 + j][v] + bi) * os;
          outf[(size_t)row * Ncols + col] = f;
          m = fmaxf(m, fabsf(f));
        }
    }
    atomicMax((unsigned*)&g[9], __float_as_uint(m));
  }
}

__global__ void __launch_bounds__(128)
k_scores_absmax(const int8_t* __restrict__ h8, float* __restrict__ g) {
  int lane = threadIdx.x & 31, wv = threadIdx.x >> 5;
  int bh = blockIdx.x / NT_S, rt = blockIdx.x % NT_S;
  int b = bh / HN, hh = bh % HN;
  const int8_t* qb = h8 + ((size_t)(b * NSEQ + rt * 16)) * NQKV + hh * DH;
  v8i a = load_a16x64(qb, NQKV, 0, lane);
  int mloc = 0;
  for (int nt = wv; nt < NT_S; nt += 4) {
    const int8_t* kb = h8 + ((size_t)(b * NSEQ + nt * 16)) * NQKV + CDIM + hh * DH;
    v8i bm = load_b64x16(kb, NQKV, 0, lane);
    v8i acc = {};
    acc = __builtin_amdgcn_wmma_i32_16x16x64_iu8(true, a, true, bm, acc, false, false);
    for (int v = 0; v < 8; v++) { int s = acc[v]; s = s < 0 ? -s : s; if (s > mloc) mloc = s; }
  }
  atomicMax((int*)&g[2], mloc);
}

__global__ void k_transpose_v(const int8_t* __restrict__ h8, int8_t* __restrict__ vT) {
  size_t i = (size_t)blockIdx.x * 256 + threadIdx.x;
  int n = (int)(i % NPAD);
  int d = (int)((i / NPAD) % DH);
  int bh = (int)(i / ((size_t)NPAD * DH));
  int b = bh / HN, hh = bh % HN;
  int8_t v = 0;
  if (n < NSEQ) v = h8[((size_t)(b * NSEQ + n)) * NQKV + 2 * CDIM + hh * DH + d];
  vT[i] = v;
}

__device__ __forceinline__ float expint(int a8, int rmax, int x0) {
  int x = a8 - rmax;
  x = x + (x >> 1) - (x >> 4);
  int nx0 = 15 * x0;
  if (x < nx0) x = nx0;
  int q = x / x0;
  int rem = x - x0 * q;
  float e = floorf(ldexpf(0.5f * (float)rem - (float)x0, 15 - q));
  return e < 0.f ? 0.f : e;
}

__global__ void __launch_bounds__(128)
k_fused_softmax_ctx(const int8_t* __restrict__ h8, const int8_t* __restrict__ vT,
                    float* __restrict__ g, int* __restrict__ ctx32) {
  __shared__ __align__(16) int8_t  sc8[16][NSEQ];
  __shared__ __align__(16) uint8_t hiA[16][NPAD];
  __shared__ __align__(16) uint8_t loA[16][NPAD];
  __shared__ int   redi[16][8];
  __shared__ float redf[16][8];
  __shared__ int   rowmax[16];
  __shared__ float rowfac[16];

  int t = threadIdx.x, lane = t & 31, wv = t >> 5;
  int bh = blockIdx.x / NT_S, rt = blockIdx.x % NT_S;
  int b = bh / HN, hh = bh % HN;
  float rscale = g[5];
  int x0 = ((const int*)g)[4];
  int half = (lane >> 4) & 1, cl = lane & 15;

  uint32_t* hz = (uint32_t*)&hiA[0][0];
  uint32_t* lz = (uint32_t*)&loA[0][0];
  for (int i = t; i < 16 * NPAD / 4; i += 128) { hz[i] = 0u; lz[i] = 0u; }

  const int8_t* qb = h8 + ((size_t)(b * NSEQ + rt * 16)) * NQKV + hh * DH;
  v8i aq = load_a16x64(qb, NQKV, 0, lane);
  for (int nt = wv; nt < NT_S; nt += 4) {
    const int8_t* kb = h8 + ((size_t)(b * NSEQ + nt * 16)) * NQKV + CDIM + hh * DH;
    v8i bm = load_b64x16(kb, NQKV, 0, lane);
    v8i acc = {};
    acc = __builtin_amdgcn_wmma_i32_16x16x64_iu8(true, aq, true, bm, acc, false, false);
    for (int v = 0; v < 8; v++)
      sc8[v + 8 * half][nt * 16 + cl] =
          (int8_t)clip_round((float)acc[v] * rscale, -128.f, 127.f);
  }
  __syncthreads();

  int r = t >> 3, j = t & 7;
  int lm = -(1 << 30);
  for (int c = j; c < NSEQ; c += 8) { int v = sc8[r][c]; if (v > lm) lm = v; }
  redi[r][j] = lm; __syncthreads();
  if (j == 0) { int m = redi[r][0]; for (int k = 1; k < 8; k++) m = max(m, redi[r][k]); rowmax[r] = m; }
  __syncthreads();
  int rm = rowmax[r];
  float ls = 0.f;
  for (int c = j; c < NSEQ; c += 8) ls += expint(sc8[r][c], rm, x0);
  redf[r][j] = ls; __syncthreads();
  if (j == 0) {
    float s = 0.f; for (int k = 0; k < 8; k++) s += redf[r][k];
    s = fminf(s, 2147483647.0f);
    rowfac[r] = floorf(2147483647.0f / s);
  }
  __syncthreads();
  float fac = rowfac[r];
  for (int c = j; c < NSEQ; c += 8) {
    float e = expint(sc8[r][c], rm, x0);
    unsigned a16 = (unsigned)fmaxf(0.f, floorf(e * fac * (1.0f / 65536.0f)));
    hiA[r][c] = (uint8_t)(a16 >> 8);
    loA[r][c] = (uint8_t)(a16 & 255u);
  }
  __syncthreads();

  const int8_t* vb = vT + ((size_t)(bh * DH + wv * 16)) * NPAD;
  v8i acch = {}, accl = {};
  for (int ks = 0; ks < NPAD / 64; ks++) {
    int k0 = ks * 64;
    v8i ah = load_a16x64((const int8_t*)&hiA[0][0], NPAD, k0, lane);
    v8i al = load_a16x64((const int8_t*)&loA[0][0], NPAD, k0, lane);
    v8i bv = load_b64x16(vb, NPAD, k0, lane);
    acch = __builtin_amdgcn_wmma_i32_16x16x64_iu8(false, ah, true, bv, acch, false, false);
    accl = __builtin_amdgcn_wmma_i32_16x16x64_iu8(false, al, true, bv, accl, false, false);
  }
  int lmax = 0;
  for (int v = 0; v < 8; v++) {
    int res = (acch[v] << 8) + accl[v];
    int row = rt * 16 + v + 8 * half;
    int col = hh * DH + wv * 16 + cl;
    ctx32[((size_t)(b * NSEQ + row)) * CDIM + col] = res;
    int s = res < 0 ? -res : res; if (s > lmax) lmax = s;
  }
  atomicMax((int*)&g[6], lmax);
}

__global__ void k_requant_ctx(const int* __restrict__ ctx32, const float* __restrict__ g,
                              int8_t* __restrict__ ctx8) {
  size_t i = (size_t)blockIdx.x * 256 + threadIdx.x;
  ctx8[i] = (int8_t)clip_round((float)ctx32[i] * g[7], -128.f, 127.f);
}

__global__ void k_scalars(float* g, int stage, float* out_tail) {
  if (threadIdx.x != 0 || blockIdx.x != 0) return;
  if (stage == 1) {
    g[1] = g[0] / 127.0f;
  } else if (stage == 2) {
    int ai = ((int*)g)[2];
    float s1 = g[1];
    float s_attn = (float)ai * s1 * s1 * 0.125f / 127.0f;
    g[3] = s_attn;
    ((int*)g)[4] = (int)floorf(-1.0f / s_attn);
    g[5] = 127.0f / (float)ai;
  } else if (stage == 3) {
    int ci = ((int*)g)[6];
    g[7] = 127.0f / (float)ci;
    g[8] = (float)ci * g[1] * (1.0f / 32768.0f) / 127.0f;
  } else {
    g[10] = g[9] / 32767.0f;
    out_tail[0] = g[10];
  }
}

__global__ void k_final(float* __restrict__ out, const float* __restrict__ g) {
  size_t i = (size_t)blockIdx.x * 256 + threadIdx.x;
  float s_out = g[10];
  int q = clip_round(out[i] / s_out, -32768.f, 32767.f);
  out[i] = (float)q * s_out;
}

extern "C" void kernel_launch(void* const* d_in, const int* in_sizes, int n_in,
                              void* d_out, int out_size, void* d_ws, size_t ws_size,
                              hipStream_t stream) {
  (void)in_sizes; (void)n_in; (void)ws_size;
  const float* x    = (const float*)d_in[0];
  const float* sx   = (const float*)d_in[1];
  const float* Wq   = (const float*)d_in[2];
  const float* Wp   = (const float*)d_in[3];
  const float* bp   = (const float*)d_in[4];
  float* out = (float*)d_out;

  char* w = (char*)d_ws;
  size_t o = 0;
  auto alloc = [&](size_t bytes) { char* p = w + o; o = (o + bytes + 255) & ~(size_t)255; return p; };
  float*  g    = (float*)alloc(1024);
  int8_t* x8   = (int8_t*)alloc((size_t)MROWS * CDIM);
  int8_t* wq8  = (int8_t*)alloc((size_t)NQKV * CDIM);
  int8_t* wp8  = (int8_t*)alloc((size_t)CDIM * CDIM);
  float*  wqs  = (float*)alloc((size_t)NQKV * 4);
  float*  wps  = (float*)alloc((size_t)CDIM * 4);
  int8_t* h8   = (int8_t*)alloc((size_t)MROWS * NQKV);
  int8_t* vT   = (int8_t*)alloc((size_t)BSZ * HN * DH * NPAD);
  int*    ctx32= (int*)alloc((size_t)MROWS * CDIM * 4);
  int8_t* ctx8 = (int8_t*)alloc((size_t)MROWS * CDIM);

  k_init<<<1, 32, 0, stream>>>(g);
  k_quant_x<<<(MROWS * CDIM) / 256, 256, 0, stream>>>(x, sx, x8);
  k_quant_w<<<NQKV + CDIM, 256, 0, stream>>>(Wq, Wp, wq8, wp8, wqs, wps);

  k_gemm_i8<0><<<MB_X * NB_Q, 128, 0, stream>>>(x8, wq8, NB_Q, NQKV, CDIM,
                                                wqs, sx, g, nullptr, nullptr, nullptr);
  k_scalars<<<1, 1, 0, stream>>>(g, 1, nullptr);
  k_gemm_i8<1><<<MB_X * NB_Q, 128, 0, stream>>>(x8, wq8, NB_Q, NQKV, CDIM,
                                                wqs, sx, g, h8, nullptr, nullptr);

  k_scores_absmax<<<BSZ * HN * NT_S, 128, 0, stream>>>(h8, g);
  k_scalars<<<1, 1, 0, stream>>>(g, 2, nullptr);

  k_transpose_v<<<(BSZ * HN * DH * NPAD) / 256, 256, 0, stream>>>(h8, vT);
  k_fused_softmax_ctx<<<BSZ * HN * NT_S, 128, 0, stream>>>(h8, vT, g, ctx32);
  k_scalars<<<1, 1, 0, stream>>>(g, 3, nullptr);

  k_requant_ctx<<<(MROWS * CDIM) / 256, 256, 0, stream>>>(ctx32, g, ctx8);
  k_gemm_i8<2><<<MB_X * NB_P, 128, 0, stream>>>(ctx8, wp8, NB_P, CDIM, CDIM,
                                                wps, sx, g, nullptr, out, bp);

  k_scalars<<<1, 1, 0, stream>>>(g, 4, out + (out_size - 1));
  k_final<<<(MROWS * CDIM) / 256, 256, 0, stream>>>(out, g);
}
